// APRConvNet_49984829391028
// MI455X (gfx1250) — matrix operand order repaired and checked by reading, not compiled, not run on hardware
//
#include <hip/hip_runtime.h>
#include <hip/hip_bf16.h>

#define B_   4
#define NP   1000000
#define NP1  125000
#define NP2  15625
#define KNB  27
#define C1   32
#define C2   64
#define T1   62500     // NP/16 (exact)
#define T2   7813      // ceil(NP1/16)
#define T3   977       // ceil(NP2/16)
#define SBLK 128       // fixed #blocks for deterministic stats partials

typedef __attribute__((ext_vector_type(16))) __bf16 v16bf;
typedef __attribute__((ext_vector_type(8)))  __bf16 v8bf;
typedef __attribute__((ext_vector_type(8)))  float  v8f;

__device__ __forceinline__ v8f wmma_bf16(v16bf a, v16bf b, v8f c) {
  // D = A(16x32 bf16) * B(32x16 bf16) + C(16x16 f32)
  return __builtin_amdgcn_wmma_f32_16x16x32_bf16(false, a, false, b, (short)0, c, false, false);
}

// A-fragment: lane row = lane&15; element j -> K=16h+j.
// rowK points at 32 K-contiguous bf16 for this output row.
__device__ __forceinline__ v16bf load_afrag(const __bf16* rowK, int h) {
  v8bf lo = *(const v8bf*)(rowK + 16 * h);
  v8bf hi = *(const v8bf*)(rowK + 16 * h + 8);
  v16bf r;
#pragma unroll
  for (int i = 0; i < 8; i++) { r[i] = lo[i]; r[i + 8] = hi[i]; }
  return r;
}

// Fused ReLU + segment-max: values clamped >=0, buffer init to 0, so int-bit
// atomicMax is monotonic/correct and bitwise deterministic.
__device__ __forceinline__ void amax_relu(float* p, float v) {
  v = v > 0.f ? v : 0.f;
  atomicMax((int*)p, __float_as_int(v));
}

__global__ void k_zero(float* __restrict__ p, size_t n) {
  size_t i = (size_t)blockIdx.x * blockDim.x + threadIdx.x;
  size_t s = (size_t)gridDim.x * blockDim.x;
  for (; i < n; i += s) p[i] = 0.f;
}

// parts [4][NP] -> xT [NP][4] so a neighbor gather reads one 16B float4.
__global__ void k_transpose(const float* __restrict__ parts, float* __restrict__ xT) {
  int n = blockIdx.x * blockDim.x + threadIdx.x;
  if (n >= NP) return;
  float4 v;
  v.x = parts[0 * (size_t)NP + n]; v.y = parts[1 * (size_t)NP + n];
  v.z = parts[2 * (size_t)NP + n]; v.w = parts[3 * (size_t)NP + n];
  *(float4*)(xT + 4 * (size_t)n) = v;
}

// bf16 weight copies, K-contiguous per output row.
// w1b [2][32][32] (k padded 27->32 with zeros), w2b [64][27][32] (K'=k*32+c, c contiguous),
// wfc1b [64][64].
__global__ void k_wprep(const float* __restrict__ w1, const float* __restrict__ w2,
                        const float* __restrict__ wfc1, __bf16* __restrict__ w1b,
                        __bf16* __restrict__ w2b, __bf16* __restrict__ wfc1b) {
  int i = blockIdx.x * blockDim.x + threadIdx.x;
  if (i < 2 * 32 * 32) {
    int k = i & 31, o = (i >> 5) & 31, s = i >> 10;
    float v = (k < KNB) ? w1[(s * 32 + o) * KNB + k] : 0.f;
    w1b[i] = (__bf16)v;
  }
  if (i < 64 * KNB * 32) {
    int c = i & 31, t = i >> 5;
    int k = t % KNB, o = t / KNB;
    w2b[i] = (__bf16)w2[(o * 32 + c) * KNB + k];
  }
  if (i < 64 * 64) wfc1b[i] = (__bf16)wfc1[i];
}

// conv1 (Cin=1, both stencils via WMMA, per-column select) fused with bias +
// segment-max-pool + ReLU via atomicMax. One wave per 16-particle tile.
__global__ void k_conv1(const float* __restrict__ xT, const int* __restrict__ nbr1,
                        const int* __restrict__ sten, const int* __restrict__ pidx,
                        const __bf16* __restrict__ w1b, const float* __restrict__ b1,
                        float* __restrict__ pool1) {
  __shared__ __bf16 sx[8][4][16][32];   // [wave][batch][n][k]
  int lane = threadIdx.x & 31, wv = threadIdx.x >> 5;
  int tile = blockIdx.x * 8 + wv;
  if (tile >= T1) tile = T1 - 1;        // duplicate work; atomicMax idempotent
  int h = lane >> 4, m = lane & 15;

  // cooperative gather of 16x27 neighbor scalars x 4 batches
  for (int i = lane; i < 16 * 32; i += 32) {
    int nr = i >> 5, k = i & 31;
    float4 v = make_float4(0.f, 0.f, 0.f, 0.f);
    if (k < KNB) {
      int nn = tile * 16 + nr;
      int idx = nbr1[(size_t)nn * KNB + k];
      v = *(const float4*)(xT + 4 * (size_t)idx);
    }
    sx[wv][0][nr][k] = (__bf16)v.x; sx[wv][1][nr][k] = (__bf16)v.y;
    sx[wv][2][nr][k] = (__bf16)v.z; sx[wv][3][nr][k] = (__bf16)v.w;
  }
  __builtin_amdgcn_wave_barrier();

  int n  = tile * 16 + m;
  int sel = sten[n];
  int p1  = pidx[n];
  v8f z = {0.f, 0.f, 0.f, 0.f, 0.f, 0.f, 0.f, 0.f};
#pragma unroll
  for (int b = 0; b < 4; b++) {
    v16bf bf = *(const v16bf*)&sx[wv][b][m][16 * h];
    v8f acc[2][2];
#pragma unroll
    for (int s = 0; s < 2; s++)
#pragma unroll
      for (int ot = 0; ot < 2; ot++) {
        v16bf af = load_afrag(w1b + (size_t)(s * 32 + ot * 16 + m) * 32, h);
        acc[s][ot] = wmma_bf16(af, bf, z);
      }
#pragma unroll
    for (int ot = 0; ot < 2; ot++) {
      v8f d = (sel == 1) ? acc[1][ot] : acc[0][ot];
#pragma unroll
      for (int r = 0; r < 8; r++) {
        int o = ot * 16 + 8 * h + r;
        amax_relu(&pool1[((size_t)b * NP1 + p1) * C1 + o], d[r] + b1[o]);
      }
    }
  }
}

// conv2: per (tile,batch) wave; 27 K-steps of 32 channels, 4 o-tiles.
// B-fragment is a direct 32B contiguous gather from the particle-major bf16 input.
__global__ void k_conv2(const __bf16* __restrict__ xn, const int* __restrict__ nbr2,
                        const int* __restrict__ pidx2, const __bf16* __restrict__ w2b,
                        const float* __restrict__ b2, float* __restrict__ pool2) {
  int lane = threadIdx.x & 31, wv = threadIdx.x >> 5;
  int w = blockIdx.x * 8 + wv;
  if (w >= T2 * 4) return;              // wave-uniform exit (no barriers here)
  int tile = w % T2, b = w / T2;
  int h = lane >> 4, m = lane & 15;
  int n = tile * 16 + m; if (n > NP1 - 1) n = NP1 - 1;
  int p2 = pidx2[n];
  const size_t xbase = (size_t)b * NP1 * C1;
  v8f acc[4];
  v8f z = {0.f, 0.f, 0.f, 0.f, 0.f, 0.f, 0.f, 0.f};
#pragma unroll
  for (int ot = 0; ot < 4; ot++) acc[ot] = z;
  for (int k = 0; k < KNB; k++) {
    int idx = nbr2[(size_t)n * KNB + k];
    v16bf bf = *(const v16bf*)(xn + xbase + (size_t)idx * C1 + 16 * h);
#pragma unroll
    for (int ot = 0; ot < 4; ot++) {
      v16bf af = load_afrag(w2b + ((size_t)(ot * 16 + m) * KNB + k) * 32, h);
      acc[ot] = wmma_bf16(af, bf, acc[ot]);
    }
  }
#pragma unroll
  for (int ot = 0; ot < 4; ot++)
#pragma unroll
    for (int r = 0; r < 8; r++) {
      int o = ot * 16 + 8 * h + r;
      amax_relu(&pool2[((size_t)b * NP2 + p2) * C2 + o], acc[ot][r] + b2[o]);
    }
}

// fc1: 64x64 GEMM per 16-particle tile, 2 K-steps, fused bias+ReLU.
__global__ void k_fc1(const __bf16* __restrict__ xn, const __bf16* __restrict__ wb,
                      const float* __restrict__ bfc1, float* __restrict__ y) {
  int lane = threadIdx.x & 31, wv = threadIdx.x >> 5;
  int w = blockIdx.x * 8 + wv;
  if (w >= T3 * 4) return;
  int tile = w % T3, b = w / T3;
  int h = lane >> 4, m = lane & 15;
  int n = tile * 16 + m; if (n > NP2 - 1) n = NP2 - 1;
  v8f acc[4];
  v8f z = {0.f, 0.f, 0.f, 0.f, 0.f, 0.f, 0.f, 0.f};
#pragma unroll
  for (int ot = 0; ot < 4; ot++) acc[ot] = z;
#pragma unroll
  for (int kk = 0; kk < 2; kk++) {
    v16bf bf = *(const v16bf*)(xn + ((size_t)b * NP2 + n) * C2 + kk * 32 + 16 * h);
#pragma unroll
    for (int ot = 0; ot < 4; ot++) {
      v16bf af = load_afrag(wb + (size_t)(ot * 16 + m) * C2 + kk * 32, h);
      acc[ot] = wmma_bf16(af, bf, acc[ot]);
    }
  }
#pragma unroll
  for (int ot = 0; ot < 4; ot++)
#pragma unroll
    for (int r = 0; r < 8; r++) {
      int o = ot * 16 + 8 * h + r;
      float v = acc[ot][r] + bfc1[o];
      y[((size_t)b * NP2 + n) * C2 + o] = v > 0.f ? v : 0.f;  // tail dupes write same value
    }
}

// Deterministic per-channel sum/sumsq: fixed SBLK partials, fixed block tree.
template <int C>
__global__ void k_stats(const float* __restrict__ x, long rows, float* __restrict__ part) {
  __shared__ float ss[256], s2[256];
  int tid = threadIdx.x;
  int c = tid % C, g = tid / C, G = 256 / C;
  float a = 0.f, aa = 0.f;
  for (long r = (long)blockIdx.x * G + g; r < rows; r += (long)gridDim.x * G) {
    float v = x[r * C + c];
    a += v; aa += v * v;
  }
  ss[tid] = a; s2[tid] = aa; __syncthreads();
  for (int st = 128; st >= C; st >>= 1) {
    if (tid < st) { ss[tid] += ss[tid + st]; s2[tid] += s2[tid + st]; }
    __syncthreads();
  }
  if (tid < C) {
    part[((long)blockIdx.x * 2 + 0) * C + tid] = ss[tid];
    part[((long)blockIdx.x * 2 + 1) * C + tid] = s2[tid];
  }
}

template <int C>
__global__ void k_stats_reduce(const float* __restrict__ part, int nblk, float* __restrict__ out) {
  int c = threadIdx.x; if (c >= C) return;
  float a = 0.f, aa = 0.f;
  for (int i = 0; i < nblk; i++) { a += part[(2 * i) * C + c]; aa += part[(2 * i + 1) * C + c]; }
  out[c] = a; out[C + c] = aa;
}

// BN (training stats) applied + bf16 particle-major write for the next GEMM stage.
template <int C>
__global__ void k_bnapply(const float* __restrict__ x, const float* __restrict__ st,
                          const float* __restrict__ g, const float* __restrict__ beta,
                          long rows, __bf16* __restrict__ y) {
  long i = (long)blockIdx.x * blockDim.x + threadIdx.x;
  if (i >= rows) return;
  float inv = 1.f / (float)rows;
#pragma unroll
  for (int c = 0; c < C; c++) {
    float mu = st[c] * inv;
    float var = st[C + c] * inv - mu * mu;
    float rs = rsqrtf(var + 1e-5f);
    float v = x[i * C + c];
    y[i * C + c] = (__bf16)(g[c] * (v - mu) * rs + beta[c]);
  }
}

// Combine per-batch fc1 stats: sum_bc[4][64] then sumsq_c[64] (serial, deterministic).
__global__ void k_fcreduce(const float* __restrict__ part, int nblk, float* __restrict__ out) {
  int c = threadIdx.x; if (c >= 64) return;
  float sq = 0.f;
  for (int b = 0; b < 4; b++) {
    float a = 0.f;
    for (int i = 0; i < nblk; i++) {
      a  += part[(((long)b * nblk + i) * 2 + 0) * 64 + c];
      sq += part[(((long)b * nblk + i) * 2 + 1) * 64 + c];
    }
    out[b * 64 + c] = a;
  }
  out[256 + c] = sq;
}

// mean∘BN∘fc2 commute: logits from per-(b,c) means; then softmax -> d_out[4][10].
__global__ void k_final(const float* __restrict__ fcs, const float* __restrict__ gfc,
                        const float* __restrict__ betafc, const float* __restrict__ wfc2,
                        const float* __restrict__ bfc2, float* __restrict__ out) {
  __shared__ float sm[4][64];
  __shared__ float lg[4][10];
  int tid = threadIdx.x;
  const float cnt = 4.f * (float)NP2;
  if (tid < 64) {
    float tot = fcs[tid] + fcs[64 + tid] + fcs[128 + tid] + fcs[192 + tid];
    float mu = tot / cnt;
    float var = fcs[256 + tid] / cnt - mu * mu;
    float rs = rsqrtf(var + 1e-5f);
    for (int b = 0; b < 4; b++) {
      float mb = fcs[b * 64 + tid] / (float)NP2;
      sm[b][tid] = gfc[tid] * (mb - mu) * rs + betafc[tid];
    }
  }
  __syncthreads();
  if (tid < 40) {
    int b = tid / 10, o = tid % 10;
    float a = bfc2[o];
    for (int c = 0; c < 64; c++) a += wfc2[o * 64 + c] * sm[b][c];
    lg[b][o] = a;
  }
  __syncthreads();
  if (tid < 4) {
    float mx = -1e30f;
    for (int o = 0; o < 10; o++) mx = fmaxf(mx, lg[tid][o]);
    float e[10], s = 0.f;
    for (int o = 0; o < 10; o++) { e[o] = __expf(lg[tid][o] - mx); s += e[o]; }
    for (int o = 0; o < 10; o++) out[tid * 10 + o] = e[o] / s;
  }
}

extern "C" void kernel_launch(void* const* d_in, const int* in_sizes, int n_in,
                              void* d_out, int out_size, void* d_ws, size_t ws_size,
                              hipStream_t stream) {
  (void)in_sizes; (void)n_in; (void)out_size; (void)ws_size;
  const float* parts = (const float*)d_in[0];
  const float* w1    = (const float*)d_in[1];
  const float* b1    = (const float*)d_in[2];
  const float* w2    = (const float*)d_in[3];
  const float* b2    = (const float*)d_in[4];
  const float* wfc1  = (const float*)d_in[5];
  const float* bfc1  = (const float*)d_in[6];
  const float* wfc2  = (const float*)d_in[7];
  const float* bfc2  = (const float*)d_in[8];
  const float* g1    = (const float*)d_in[9];
  const float* beta1 = (const float*)d_in[10];
  const float* g2    = (const float*)d_in[11];
  const float* beta2 = (const float*)d_in[12];
  const float* gfc   = (const float*)d_in[13];
  const float* betafc= (const float*)d_in[14];
  const int* nbr1    = (const int*)d_in[15];
  const int* sten1   = (const int*)d_in[16];
  const int* pidx1   = (const int*)d_in[17];
  const int* nbr2    = (const int*)d_in[18];
  const int* pidx2   = (const int*)d_in[19];
  float* out = (float*)d_out;

  char* ws = (char*)d_ws; size_t off = 0;
  auto take = [&](size_t bytes) -> char* {
    char* p = ws + off;
    off = (off + bytes + 255) & ~(size_t)255;
    return p;
  };
  float*  xT     = (float*)  take((size_t)NP * 4 * 4);
  float*  pool1  = (float*)  take((size_t)B_ * NP1 * C1 * 4);
  __bf16* xn2    = (__bf16*) take((size_t)B_ * NP1 * C1 * 2);
  float*  pool2  = (float*)  take((size_t)B_ * NP2 * C2 * 4);
  __bf16* xnfc   = (__bf16*) take((size_t)B_ * NP2 * C2 * 2);
  float*  ybuf   = (float*)  take((size_t)B_ * NP2 * C2 * 4);
  __bf16* w1b    = (__bf16*) take((size_t)2 * 32 * 32 * 2);
  __bf16* w2b    = (__bf16*) take((size_t)64 * KNB * 32 * 2);
  __bf16* wfc1b  = (__bf16*) take((size_t)64 * 64 * 2);
  float*  part1  = (float*)  take((size_t)SBLK * 2 * C1 * 4);
  float*  stats1 = (float*)  take((size_t)2 * C1 * 4);
  float*  part2  = (float*)  take((size_t)SBLK * 2 * C2 * 4);
  float*  stats2 = (float*)  take((size_t)2 * C2 * 4);
  float*  fcpart = (float*)  take((size_t)4 * SBLK * 2 * C2 * 4);
  float*  fcs    = (float*)  take((size_t)(4 * 64 + 64) * 4);

  k_zero<<<2048, 256, 0, stream>>>(pool1, (size_t)B_ * NP1 * C1);
  k_zero<<<512, 256, 0, stream>>>(pool2, (size_t)B_ * NP2 * C2);
  k_transpose<<<(NP + 255) / 256, 256, 0, stream>>>(parts, xT);
  k_wprep<<<(64 * KNB * 32 + 255) / 256, 256, 0, stream>>>(w1, w2, wfc1, w1b, w2b, wfc1b);

  k_conv1<<<(T1 + 7) / 8, 256, 0, stream>>>(xT, nbr1, sten1, pidx1, w1b, b1, pool1);
  k_stats<C1><<<SBLK, 256, 0, stream>>>(pool1, (long)B_ * NP1, part1);
  k_stats_reduce<C1><<<1, C1, 0, stream>>>(part1, SBLK, stats1);
  k_bnapply<C1><<<(int)(((long)B_ * NP1 + 255) / 256), 256, 0, stream>>>(
      pool1, stats1, g1, beta1, (long)B_ * NP1, xn2);

  k_conv2<<<(T2 * 4 + 7) / 8, 256, 0, stream>>>(xn2, nbr2, pidx2, w2b, b2, pool2);
  k_stats<C2><<<SBLK, 256, 0, stream>>>(pool2, (long)B_ * NP2, part2);
  k_stats_reduce<C2><<<1, C2, 0, stream>>>(part2, SBLK, stats2);
  k_bnapply<C2><<<(int)(((long)B_ * NP2 + 255) / 256), 256, 0, stream>>>(
      pool2, stats2, g2, beta2, (long)B_ * NP2, xnfc);

  k_fc1<<<(T3 * 4 + 7) / 8, 256, 0, stream>>>(xnfc, wfc1b, bfc1, ybuf);
  for (int b = 0; b < 4; b++)
    k_stats<C2><<<SBLK, 256, 0, stream>>>(ybuf + (size_t)b * NP2 * C2, (long)NP2,
                                          fcpart + (size_t)b * SBLK * 2 * C2);
  k_fcreduce<<<1, 64, 0, stream>>>(fcpart, SBLK, fcs);
  k_final<<<1, 64, 0, stream>>>(fcs, gfc, betafc, wfc2, bfc2, out);
}
